// MPNNModel_89970974917473
// MI455X (gfx1250) — matrix operand order repaired and checked by reading, not compiled, not run on hardware
//
#include <hip/hip_runtime.h>

// ---------------------------------------------------------------------------
// MPNN forward, restructured for MI455X (gfx1250, wave32, WMMA):
//   h = x @ Wp + bp
//   per layer l:
//     P    = h @ W1a[l] + b1[l]                    (node GEMM, WMMA bf16)
//     S    = scatter_add_dst( relu(P[src] + edge_attr @ W1b[l]) )
//     aggr = S @ W2[l] + deg * b2[l]               (node GEMM, WMMA bf16)
//     u    = h @ U1[l] + aggr @ U2[l] + ub[l]      (two node GEMMs, WMMA)
//     h    = relu(LN(relu(u))) + h
//   pooling (mean/max per graph) + 2-layer readout FFN.
// Weights are pre-transposed/converted to bf16 once per call; GEMM blocks pull
// them into LDS via gfx1250 async-to-LDS (global_load_async_to_lds_b128).
// ---------------------------------------------------------------------------

typedef __bf16 bf16;
typedef __attribute__((ext_vector_type(16))) bf16  v16bf;
typedef __attribute__((ext_vector_type(8)))  bf16  v8bf;
typedef __attribute__((ext_vector_type(8)))  float v8f;
typedef __attribute__((ext_vector_type(4)))  int   v4i;

#define N_NODES 20000
#define N_EDGES 640000
#define IN_CH   64
#define EDGE_DIM 16
#define HID     128
#define N_LAYERS 3
#define N_GRAPHS 128

#if defined(__AMDGCN__) && __has_builtin(__builtin_amdgcn_global_load_async_to_lds_b128) && __has_builtin(__builtin_amdgcn_s_wait_asynccnt)
#define USE_ASYNC_LDS 1
#else
#define USE_ASYNC_LDS 0
#endif

// Cooperative copy of `bytes` (multiple of 16, per 128-thread block) from
// global bf16 weights into LDS. Async-to-LDS path on gfx1250 when available.
__device__ inline void block_copy_w(const bf16* __restrict__ gsrc, bf16* ldst,
                                    int bytes, int tid) {
#if USE_ASYNC_LDS
    __attribute__((address_space(1))) v4i* g =
        (__attribute__((address_space(1))) v4i*)gsrc;
    __attribute__((address_space(3))) v4i* l =
        (__attribute__((address_space(3))) v4i*)ldst;
    const int n = bytes >> 4;
    for (int i = tid; i < n; i += 128)
        __builtin_amdgcn_global_load_async_to_lds_b128(g + i, l + i, 0, 0);
    __builtin_amdgcn_s_wait_asynccnt(0);
#else
    const v8bf* gs = (const v8bf*)gsrc;
    v8bf* ls = (v8bf*)ldst;
    const int n = bytes >> 4;
    for (int i = tid; i < n; i += 128) ls[i] = gs[i];
#endif
}

// ---- WMMA fragment loaders (layouts per CDNA5 ISA 7.12.2, wave32) ---------

// A matrix 16x32 (bf16): lane m (m<16) holds row m, K{0..7} then K{16..23};
// lane m+16 holds row m, K{8..15} then K{24..31}.
__device__ inline v16bf load_a_frag(const bf16* base, int stride, int kbase, int lane) {
    int row = lane & 15;
    int hi  = lane >> 4;
    const bf16* p = base + row * stride + kbase + hi * 8;
    union { v16bf v; v8bf h[2]; } u;
    u.h[0] = *(const v8bf*)p;        // K = kbase + hi*8 .. +7
    u.h[1] = *(const v8bf*)(p + 16); // K = kbase + 16 + hi*8 .. +7
    return u.v;
}

// B matrix 32x16 (bf16), weights stored TRANSPOSED: wt[col][k].
// lane n (n<16) holds col n, K{0..7} then K{16..23}; lane n+16 holds col n, K{8..15} then K{24..31}.
__device__ inline v16bf load_b_frag(const bf16* wt, int kstride, int colbase, int kbase, int lane) {
    int col = colbase + (lane & 15);
    int hi  = lane >> 4;
    const bf16* p = wt + col * kstride + kbase + hi * 8;
    union { v16bf v; v8bf h[2]; } u;
    u.h[0] = *(const v8bf*)p;
    u.h[1] = *(const v8bf*)(p + 16);
    return u.v;
}

// ---- weight prep: WT[n][k] = bf16(W[k][n]), K zero-padded to Kpad ---------
__global__ void transpose_cvt_kernel(const float* __restrict__ W, bf16* __restrict__ WT,
                                     int K, int Kpad) {
    int i = blockIdx.x * blockDim.x + threadIdx.x;
    int stride = gridDim.x * blockDim.x;
    const int total = HID * Kpad;
    for (; i < total; i += stride) {
        int n = i / Kpad, k = i - n * Kpad;
        WT[i] = (k < K) ? (bf16)W[k * HID + n] : (bf16)0.0f;
    }
}

// ---- generic node-level GEMM: OUT[M,128] (op)= IN[M,Kin] @ W[Kin,128] -----
// WT is pre-transposed bf16 [128][Kin].
#define GF_BIAS 1
#define GF_RELU 2
#define GF_ACC  4

__global__ __launch_bounds__(128)
void gemm_node_kernel(const float* __restrict__ IN, const bf16* __restrict__ WT,
                      const float* __restrict__ bias, const float* __restrict__ degScale,
                      float* __restrict__ OUT, int M, int Kin, int flags) {
    __shared__ __align__(16) bf16 sWT[HID * HID];     // [col][k], k < Kin
    __shared__ __align__(16) bf16 sA[4][16 * HID];    // per-wave A staging
    __shared__ float sBias[HID];

    const int tid = threadIdx.x;

    // weights: straight bf16 block copy (async-to-LDS on gfx1250)
    block_copy_w(WT, sWT, HID * Kin * (int)sizeof(bf16), tid);
    if ((flags & GF_BIAS) && tid < HID) sBias[tid] = bias[tid];

    const int wave = tid >> 5, lane = tid & 31;
    const int rowBase = blockIdx.x * 64 + wave * 16;

    // stage 16 rows of A as bf16
    bf16* A = sA[wave];
    for (int i = lane; i < 16 * Kin; i += 32) {
        int r = i / Kin, c = i - r * Kin;
        int gr = rowBase + r;
        A[i] = (gr < M) ? (bf16)IN[(long)gr * Kin + c] : (bf16)0.0f;
    }
    __syncthreads();

    const int colLo = lane & 15, rhi = (lane >> 4) * 8;
    for (int nt = 0; nt < 8; ++nt) {
        v8f acc = {};
        for (int kc = 0; kc < Kin; kc += 32) {
            v16bf a = load_a_frag(A, Kin, kc, lane);
            v16bf b = load_b_frag(sWT, Kin, nt * 16, kc, lane);
            acc = __builtin_amdgcn_wmma_f32_16x16x32_bf16(
                false, a, false, b, (short)0, acc, false, false);
        }
        const int col = nt * 16 + colLo;
        const float bcol = (flags & GF_BIAS) ? sBias[col] : 0.0f;
        for (int r = 0; r < 8; ++r) {
            int gr = rowBase + rhi + r;
            if (gr < M) {
                float v = acc[r];
                v += degScale ? degScale[gr] * bcol : bcol;
                if (flags & GF_ACC) v += OUT[(long)gr * HID + col];
                if (flags & GF_RELU) v = fmaxf(v, 0.0f);
                OUT[(long)gr * HID + col] = v;
            }
        }
    }
}

// ---- fused edge pass: S[dst] += relu(P[src] + edge_attr @ W1b) ------------
// W1bT: pre-transposed bf16 [128][32] (K padded 16 -> 32 with zeros).
__global__ __launch_bounds__(128)
void edge_msg_kernel(const int* __restrict__ ei, const float* __restrict__ eattr,
                     const float* __restrict__ P, const bf16* __restrict__ W1bT,
                     float* __restrict__ S, int E) {
    __shared__ __align__(16) bf16 sWT[HID * 32];
    __shared__ int sIdx[4][32];                   // per wave: [0..15]=src [16..31]=dst

    const int tid = threadIdx.x;
    block_copy_w(W1bT, sWT, HID * 32 * (int)sizeof(bf16), tid);

    const int wave = tid >> 5, lane = tid & 31;
    const long tile = (long)blockIdx.x * 4 + wave;
    const int e0 = (int)(tile * 16);
    const int* srcI = ei;
    const int* dstI = ei + E;
    if (lane < 16) {
        int e = e0 + lane;
        int sN = (e < E) ? srcI[e] : 0;
        sIdx[wave][lane]      = sN;
        sIdx[wave][16 + lane] = (e < E) ? dstI[e] : 0;
        // prefetch the gathered P row (512B) ahead of the WMMA
        const float* pr = P + (long)sN * HID;
        __builtin_prefetch(pr, 0, 1);
        __builtin_prefetch(pr + 64, 0, 1);
    }
    __syncthreads();

    // A fragment from edge_attr (K = 16 real + 16 zero pad)
    const int row = lane & 15, hi = lane >> 4;
    const int e = e0 + row;
    union { v16bf v; v8bf h[2]; } ua;
    const float* ap = eattr + (long)e * EDGE_DIM + hi * 8;
    const bool ok = (e < E);
    for (int j = 0; j < 8; ++j)  ua.v[j] = ok ? (bf16)ap[j] : (bf16)0.0f;
    for (int j = 8; j < 16; ++j) ua.v[j] = (bf16)0.0f;
    const v16bf a = ua.v;

    const int colLo = lane & 15;
    for (int nt = 0; nt < 8; ++nt) {
        v16bf b = load_b_frag(sWT, 32, nt * 16, 0, lane);
        v8f acc = {};
        acc = __builtin_amdgcn_wmma_f32_16x16x32_bf16(
            false, a, false, b, (short)0, acc, false, false);
        const int col = nt * 16 + colLo;
        for (int r = 0; r < 8; ++r) {
            int rr = hi * 8 + r;
            int ee = e0 + rr;
            if (ee < E) {
                int sN = sIdx[wave][rr];
                int dN = sIdx[wave][16 + rr];
                float v = acc[r] + P[(long)sN * HID + col];
                v = fmaxf(v, 0.0f);
                atomicAdd(&S[(long)dN * HID + col], v);
            }
        }
    }
}

// ---- small helper kernels -------------------------------------------------
__global__ void zero_kernel(float* p, long n) {
    long i = (long)blockIdx.x * blockDim.x + threadIdx.x;
    long stride = (long)gridDim.x * blockDim.x;
    for (; i < n; i += stride) p[i] = 0.0f;
}

__global__ void deg_kernel(const int* __restrict__ dst, float* __restrict__ deg, int E) {
    long i = (long)blockIdx.x * blockDim.x + threadIdx.x;
    long stride = (long)gridDim.x * blockDim.x;
    for (; i < E; i += stride) atomicAdd(&deg[dst[i]], 1.0f);
}

__global__ __launch_bounds__(128)
void ln_residual_kernel(const float* __restrict__ U, float* __restrict__ H,
                        const float* __restrict__ g, const float* __restrict__ b, int M) {
    __shared__ float s1[HID];
    __shared__ float s2[HID];
    const int n = blockIdx.x, c = threadIdx.x;
    float v = fmaxf(U[(long)n * HID + c], 0.0f);
    s1[c] = v; s2[c] = v * v;
    __syncthreads();
    for (int off = 64; off > 0; off >>= 1) {
        if (c < off) { s1[c] += s1[c + off]; s2[c] += s2[c + off]; }
        __syncthreads();
    }
    const float mu  = s1[0] * (1.0f / HID);
    const float var = s2[0] * (1.0f / HID) - mu * mu;
    float y = (v - mu) * rsqrtf(var + 1e-5f) * g[c] + b[c];
    H[(long)n * HID + c] = fmaxf(y, 0.0f) + H[(long)n * HID + c];
}

__global__ void init_pool_kernel(float* msum, float* mmax, float* cnt) {
    int i = blockIdx.x * blockDim.x + threadIdx.x;
    if (i < N_GRAPHS * HID) { msum[i] = 0.0f; mmax[i] = -3.0e38f; }
    if (i < N_GRAPHS) cnt[i] = 0.0f;
}

__device__ inline void atomicMaxF(float* addr, float val) {
    int* ai = (int*)addr;
    int old = __float_as_int(*addr);
    while (val > __int_as_float(old)) {
        int assumed = old;
        old = atomicCAS(ai, assumed, __float_as_int(val));
        if (old == assumed) break;
    }
}

__global__ __launch_bounds__(128)
void pool_kernel(const float* __restrict__ H, const int* __restrict__ batch,
                 float* __restrict__ msum, float* __restrict__ mmax,
                 float* __restrict__ cnt) {
    const int n = blockIdx.x, c = threadIdx.x;
    const int g = batch[n];
    const float v = H[(long)n * HID + c];
    atomicAdd(&msum[g * HID + c], v);
    atomicMaxF(&mmax[g * HID + c], v);
    if (c == 0) atomicAdd(&cnt[g], 1.0f);
}

__global__ __launch_bounds__(128)
void readout_kernel(const float* __restrict__ msum, const float* __restrict__ mmax,
                    const float* __restrict__ cnt,
                    const float* __restrict__ W1, const float* __restrict__ b1,
                    const float* __restrict__ W2, const float* __restrict__ b2,
                    float* __restrict__ out) {
    __shared__ float gv[2 * HID];
    __shared__ float red[HID];
    const int g = blockIdx.x, j = threadIdx.x;
    const float c = fmaxf(cnt[g], 1.0f);
    gv[j]        = msum[g * HID + j] / c;
    gv[HID + j]  = mmax[g * HID + j];
    __syncthreads();
    float acc = b1[j];
    for (int k = 0; k < 2 * HID; ++k) acc += gv[k] * W1[k * HID + j];
    acc = fmaxf(acc, 0.0f);
    red[j] = acc * W2[j];
    __syncthreads();
    for (int off = 64; off > 0; off >>= 1) {
        if (j < off) red[j] += red[j + off];
        __syncthreads();
    }
    if (j == 0) out[g] = red[0] + b2[0];
}

// ---------------------------------------------------------------------------
extern "C" void kernel_launch(void* const* d_in, const int* in_sizes, int n_in,
                              void* d_out, int out_size, void* d_ws, size_t ws_size,
                              hipStream_t stream) {
    const float* x       = (const float*)d_in[0];
    const int*   ei      = (const int*)  d_in[1];   // [2][E]
    const float* eattr   = (const float*)d_in[2];
    const int*   batch   = (const int*)  d_in[3];
    const float* npw     = (const float*)d_in[4];
    const float* npb     = (const float*)d_in[5];
    const float* mw1     = (const float*)d_in[6];   // [L][144][128]
    const float* mb1     = (const float*)d_in[7];
    const float* mw2     = (const float*)d_in[8];   // [L][128][128]
    const float* mb2     = (const float*)d_in[9];
    const float* uw      = (const float*)d_in[10];  // [L][256][128]
    const float* ub      = (const float*)d_in[11];
    const float* lng     = (const float*)d_in[12];
    const float* lnb     = (const float*)d_in[13];
    const float* fw1     = (const float*)d_in[14];
    const float* fb1     = (const float*)d_in[15];
    const float* fw2     = (const float*)d_in[16];
    const float* fb2     = (const float*)d_in[17];
    float* out = (float*)d_out;

    // workspace layout: f32 region then bf16 weight region
    const long NH = (long)N_NODES * HID;   // 2,560,000
    float* h    = (float*)d_ws;
    float* uP   = h    + NH;               // P, then u_pre (reused)
    float* S    = uP   + NH;
    float* aggr = S    + NH;
    float* deg  = aggr + NH;
    float* msum = deg  + N_NODES;
    float* mmax = msum + N_GRAPHS * HID;
    float* cnt  = mmax + N_GRAPHS * HID;

    bf16* projWT = (bf16*)(cnt + N_GRAPHS);      // [128][64]
    const long LWT = 16384L * 4 + 4096;          // per-layer bf16 elements
    bf16* lw[N_LAYERS][5];                       // W1aWT, W1bWT, W2WT, U1WT, U2WT
    for (int l = 0; l < N_LAYERS; ++l) {
        bf16* base = projWT + 128 * IN_CH + l * LWT;
        lw[l][0] = base;                 // W1aWT [128][128]
        lw[l][1] = base + 16384;         // W1bWT [128][32]
        lw[l][2] = base + 16384 + 4096;  // W2WT  [128][128]
        lw[l][3] = base + 32768 + 4096;  // U1WT  [128][128]
        lw[l][4] = base + 49152 + 4096;  // U2WT  [128][128]
    }

    const dim3 blk128(128);
    const int gemmBlocks = (N_NODES + 63) / 64;    // 313
    const int edgeBlocks = N_EDGES / (4 * 16);     // 10000

    // ---- weight prep (bf16, transposed) ----
    transpose_cvt_kernel<<<32, 256, 0, stream>>>(npw, projWT, IN_CH, IN_CH);
    for (int l = 0; l < N_LAYERS; ++l) {
        const float* W1  = mw1 + (long)l * (HID + EDGE_DIM) * HID;
        const float* W1b = W1 + (long)HID * HID;
        transpose_cvt_kernel<<<64, 256, 0, stream>>>(W1,  lw[l][0], HID, HID);
        transpose_cvt_kernel<<<16, 256, 0, stream>>>(W1b, lw[l][1], EDGE_DIM, 32);
        transpose_cvt_kernel<<<64, 256, 0, stream>>>(mw2 + (long)l * HID * HID, lw[l][2], HID, HID);
        const float* U1 = uw + (long)l * (2 * HID) * HID;
        transpose_cvt_kernel<<<64, 256, 0, stream>>>(U1,             lw[l][3], HID, HID);
        transpose_cvt_kernel<<<64, 256, 0, stream>>>(U1 + HID * HID, lw[l][4], HID, HID);
    }

    // degrees (layer-invariant)
    zero_kernel<<<256, 256, 0, stream>>>(deg, N_NODES);
    deg_kernel<<<2500, 256, 0, stream>>>(ei + N_EDGES, deg, N_EDGES);

    // h = x @ Wp + bp
    gemm_node_kernel<<<gemmBlocks, blk128, 0, stream>>>(
        x, projWT, npb, nullptr, h, N_NODES, IN_CH, GF_BIAS);

    for (int l = 0; l < N_LAYERS; ++l) {
        // P = h @ W1a + b1
        gemm_node_kernel<<<gemmBlocks, blk128, 0, stream>>>(
            h, lw[l][0], mb1 + l * HID, nullptr, uP, N_NODES, HID, GF_BIAS);

        // S = 0; S[dst] += relu(P[src] + edge_attr @ W1b)
        zero_kernel<<<2048, 256, 0, stream>>>(S, NH);
        edge_msg_kernel<<<edgeBlocks, blk128, 0, stream>>>(
            ei, eattr, uP, lw[l][1], S, N_EDGES);

        // aggr = S @ W2 + deg * b2
        gemm_node_kernel<<<gemmBlocks, blk128, 0, stream>>>(
            S, lw[l][2], mb2 + l * HID, deg, aggr, N_NODES, HID, GF_BIAS);

        // u_pre = h @ U1 + ub ; u_pre += aggr @ U2
        gemm_node_kernel<<<gemmBlocks, blk128, 0, stream>>>(
            h, lw[l][3], ub + l * HID, nullptr, uP, N_NODES, HID, GF_BIAS);
        gemm_node_kernel<<<gemmBlocks, blk128, 0, stream>>>(
            aggr, lw[l][4], nullptr, nullptr, uP, N_NODES, HID, GF_ACC);

        // h = relu(LN(relu(u_pre))) + h
        ln_residual_kernel<<<N_NODES, blk128, 0, stream>>>(
            uP, h, lng + l * HID, lnb + l * HID, N_NODES);
    }

    // pooling + readout
    init_pool_kernel<<<(N_GRAPHS * HID + 255) / 256, 256, 0, stream>>>(msum, mmax, cnt);
    pool_kernel<<<N_NODES, blk128, 0, stream>>>(h, batch, msum, mmax, cnt);
    readout_kernel<<<N_GRAPHS, blk128, 0, stream>>>(
        msum, mmax, cnt, fw1, fb1, fw2, fb2, out);

    (void)in_sizes; (void)n_in; (void)out_size; (void)ws_size;
}
